// Attention_11132555231610
// MI455X (gfx1250) — matrix operand order repaired and checked by reading, not compiled, not run on hardware
//
#include <hip/hip_runtime.h>

typedef _Float16 f16;
typedef __attribute__((ext_vector_type(16))) _Float16 v16h;
typedef __attribute__((ext_vector_type(8)))  float    v8f;

// Problem constants
#define BATCH 16
#define SEQ   1024
#define CDIM  768
#define NHEAD 12
#define HDIM  64

// ---------------------------------------------------------------------------
// WMMA fragment helpers (CDNA5 16x16x32 f16, wave32 layouts per ISA 7.12.2)
// ---------------------------------------------------------------------------

// A-matrix 16x32 f16 fragment from a row-major [16 x ld] tile.
__device__ inline v16h load_A(const f16* __restrict__ p, int ld) {
  const int lane = threadIdx.x & 31;
  const int m = lane & 15, hf = lane >> 4;
  v16h a;
#pragma unroll
  for (int v = 0; v < 8; ++v) {
    const int k = ((v & 4) << 2) + hf * 8 + ((v & 3) << 1);
    a[2 * v]     = p[m * ld + k];
    a[2 * v + 1] = p[m * ld + k + 1];
  }
  return a;
}

// B-matrix 32x16 f16 fragment from an N-major [16(n) x ld(k)] tile.
__device__ inline v16h load_B(const f16* __restrict__ p, int ld) {
  const int lane = threadIdx.x & 31;
  const int n = lane & 15, hf = lane >> 4;
  v16h b;
#pragma unroll
  for (int v = 0; v < 8; ++v) {
    const int k = ((v & 4) << 2) + hf * 8 + ((v & 3) << 1);
    b[2 * v]     = p[n * ld + k];
    b[2 * v + 1] = p[n * ld + k + 1];
  }
  return b;
}

__device__ inline v8f wmma16(v16h a, v16h b, v8f c) {
  return __builtin_amdgcn_wmma_f32_16x16x32_f16(false, a, false, b, (short)0,
                                                c, false, false);
}

// Async global -> LDS copy (16B per lane), tracked with ASYNCcnt.
__device__ inline void async_copy_b128(const void* gsrc, void* lds_dst) {
  const unsigned dst = (unsigned)(size_t)lds_dst;  // flat LDS addr = low 32b
  asm volatile("global_load_async_to_lds_b128 %0, %1, off"
               :: "v"(dst), "v"(gsrc) : "memory");
}

// ---------------------------------------------------------------------------
// Kernel 1: QKV projection.  [16384 x 768] f32  x  [2304 x 768]^T f32
// 256x128 block tile, 8 waves as 4(M)x2(N), each wave 64x64 (4x4 WMMA tiles,
// 16 WMMAs per 8 fragment loads).  Scatters f16: q (pre-scaled 1/8) & k as
// [B,H,N,64]; v transposed as [B,H,64,N].
// ---------------------------------------------------------------------------
__global__ __launch_bounds__(256) void qkv_kernel(
    const float* __restrict__ x, const float* __restrict__ wqkv,
    f16* __restrict__ qo, f16* __restrict__ ko, f16* __restrict__ vo) {
  __shared__ f16 As[256][40];
  __shared__ f16 Bs[128][40];

  const int tid = threadIdx.x;
  const int rowBase = blockIdx.y * 256;
  const int colBase = blockIdx.x * 128;
  const int wave = tid >> 5;
  const int waveM = wave & 3;   // 4 groups of 64 rows
  const int waveN = wave >> 2;  // 2 groups of 64 cols

  v8f acc[4][4] = {};

  for (int kt = 0; kt < 24; ++kt) {
    const int k0 = kt * 32;
    __syncthreads();
#pragma unroll
    for (int i = 0; i < 8; ++i) {  // A: 256x32 f32 = 2048 float4
      const int idx = tid + i * 256;
      const int row = idx >> 3, c4 = (idx & 7) << 2;
      float4 fa = *(const float4*)(x + (size_t)(rowBase + row) * CDIM + k0 + c4);
      As[row][c4 + 0] = (f16)fa.x; As[row][c4 + 1] = (f16)fa.y;
      As[row][c4 + 2] = (f16)fa.z; As[row][c4 + 3] = (f16)fa.w;
    }
#pragma unroll
    for (int i = 0; i < 4; ++i) {  // B: 128x32 f32
      const int idx = tid + i * 256;
      const int row = idx >> 3, c4 = (idx & 7) << 2;
      float4 fb = *(const float4*)(wqkv + (size_t)(colBase + row) * CDIM + k0 + c4);
      Bs[row][c4 + 0] = (f16)fb.x; Bs[row][c4 + 1] = (f16)fb.y;
      Bs[row][c4 + 2] = (f16)fb.z; Bs[row][c4 + 3] = (f16)fb.w;
    }
    __syncthreads();

    v16h a[4], b[4];
#pragma unroll
    for (int mt = 0; mt < 4; ++mt)
      a[mt] = load_A(&As[waveM * 64 + mt * 16][0], 40);
#pragma unroll
    for (int nt = 0; nt < 4; ++nt)
      b[nt] = load_B(&Bs[waveN * 64 + nt * 16][0], 40);
#pragma unroll
    for (int mt = 0; mt < 4; ++mt)
#pragma unroll
      for (int nt = 0; nt < 4; ++nt)
        acc[mt][nt] = wmma16(a[mt], b[nt], acc[mt][nt]);
  }

  const int lane = tid & 31, n16 = lane & 15, hf = lane >> 4;
#pragma unroll
  for (int mt = 0; mt < 4; ++mt)
#pragma unroll
    for (int nt = 0; nt < 4; ++nt)
#pragma unroll
      for (int r = 0; r < 8; ++r) {
        const int rowg = rowBase + waveM * 64 + mt * 16 + r + hf * 8;
        const int o = colBase + waveN * 64 + nt * 16 + n16;  // 0..2303
        const float val = acc[mt][nt][r];
        const int bidx = rowg >> 10, tok = rowg & 1023;
        const int part = o / CDIM;
        const int hh = (o % CDIM) >> 6, d = o & 63;
        const size_t hb = (size_t)bidx * NHEAD + hh;
        if (part == 0)
          qo[(hb * SEQ + tok) * HDIM + d] = (f16)(val * 0.125f);  // 1/sqrt(64)
        else if (part == 1)
          ko[(hb * SEQ + tok) * HDIM + d] = (f16)val;
        else
          vo[(hb * HDIM + d) * SEQ + tok] = (f16)val;  // transposed
      }
}

// ---------------------------------------------------------------------------
// Kernel 2: flash attention per (b,h).  grid = (SEQ/128, B*H), 8 waves,
// each wave owns 16 query rows; 128-key chunks, double-buffered via
// global_load_async_to_lds_b128 (ASYNCcnt), online softmax.
// ---------------------------------------------------------------------------
__global__ __launch_bounds__(256) void attn_kernel(
    const f16* __restrict__ q, const f16* __restrict__ kg,
    const f16* __restrict__ vg, f16* __restrict__ obuf) {
  __shared__ f16 Ks[2][128][72];    // keys x d      (144B stride, 16B mult)
  __shared__ f16 Vts[2][64][136];   // d x keys      (272B stride, 16B mult)
  __shared__ f16 Ps[8][16][136];    // per-wave 16x128 probabilities

  const int tid = threadIdx.x;
  const int bh = blockIdx.y;  // 0..191
  const int bb = bh / NHEAD, hh = bh % NHEAD;
  const int wave = tid >> 5, lane = tid & 31;
  const int n16 = lane & 15, hf = lane >> 4;
  const int qrow0 = blockIdx.x * 128 + wave * 16;

  const f16* Qp = q + ((size_t)bh * SEQ + qrow0) * HDIM;
  const f16* Kp = kg + (size_t)bh * SEQ * HDIM;
  const f16* Vp = vg + (size_t)bh * HDIM * SEQ;

  // Stage one 128-key chunk: 8 async b128 instructions per wave.
  auto stage = [&](int kb, int buf) {
#pragma unroll
    for (int t = 0; t < 4; ++t) {
      const int fl = tid + t * 256;
      {  // K chunk: 128 x 64 f16
        const int row = fl >> 3, c8 = (fl & 7) << 3;
        async_copy_b128(Kp + (size_t)(kb + row) * HDIM + c8, &Ks[buf][row][c8]);
      }
      {  // V^T chunk: 64 x 128 f16
        const int dd = fl >> 4, c8 = (fl & 15) << 3;
        async_copy_b128(Vp + (size_t)dd * SEQ + kb + c8, &Vts[buf][dd][c8]);
      }
    }
  };

  v16h aq[2];
  aq[0] = load_A(Qp, HDIM);
  aq[1] = load_A(Qp + 32, HDIM);

  float row_m[8], row_l[8];
  v8f oacc[4] = {};
#pragma unroll
  for (int r = 0; r < 8; ++r) { row_m[r] = -1e30f; row_l[r] = 0.f; }

  stage(0, 0);  // prologue: chunk 0 -> buffer 0

  const int NCHUNK = SEQ / 128;
  for (int jc = 0; jc < NCHUNK; ++jc) {
    const int buf = jc & 1;
    if (jc + 1 < NCHUNK) {
      __syncthreads();  // WAR: everyone done reading buf^1 (iteration jc-1)
      stage((jc + 1) * 128, buf ^ 1);
      // own 8 ops of chunk jc complete (async loads retire in order)
      asm volatile("s_wait_asynccnt 0x8" ::: "memory");
    } else {
      asm volatile("s_wait_asynccnt 0x0" ::: "memory");
    }
    __syncthreads();  // all waves' chunk-jc copies visible in LDS

    // S = Q * K^T  (16 x 128, 8 tiles)
    v8f s[8] = {};
#pragma unroll
    for (int nt = 0; nt < 8; ++nt)
#pragma unroll
      for (int ks = 0; ks < 2; ++ks) {
        v16h bK = load_B(&Ks[buf][nt * 16][ks * 32], 72);
        s[nt] = wmma16(aq[ks], bK, s[nt]);
      }

    // Online softmax: row max over 8 tiles + 16-lane shuffle reduce.
    float alpha[8], rs[8];
#pragma unroll
    for (int r = 0; r < 8; ++r) {
      float m = s[0][r];
#pragma unroll
      for (int nt = 1; nt < 8; ++nt) m = fmaxf(m, s[nt][r]);
      m = fmaxf(m, __shfl_xor(m, 1, 32));
      m = fmaxf(m, __shfl_xor(m, 2, 32));
      m = fmaxf(m, __shfl_xor(m, 4, 32));
      m = fmaxf(m, __shfl_xor(m, 8, 32));
      const float mnew = fmaxf(row_m[r], m);
      alpha[r] = __expf(row_m[r] - mnew);
      row_m[r] = mnew;
      rs[r] = 0.f;
    }
    // P = exp(S - m) -> per-wave LDS (C-layout -> A-fragment relayout).
#pragma unroll
    for (int nt = 0; nt < 8; ++nt)
#pragma unroll
      for (int r = 0; r < 8; ++r) {
        const float p = __expf(s[nt][r] - row_m[r]);
        rs[r] += p;
        Ps[wave][r + hf * 8][nt * 16 + n16] = (f16)p;
      }
#pragma unroll
    for (int r = 0; r < 8; ++r) {
      float t = rs[r];
      t += __shfl_xor(t, 1, 32);
      t += __shfl_xor(t, 2, 32);
      t += __shfl_xor(t, 4, 32);
      t += __shfl_xor(t, 8, 32);
      row_l[r] = row_l[r] * alpha[r] + t;
    }
#pragma unroll
    for (int nd = 0; nd < 4; ++nd)
#pragma unroll
      for (int r = 0; r < 8; ++r) oacc[nd][r] *= alpha[r];

    // O += P * V   (16x128 x 128x64); within-wave DS ordering covers Ps.
    v16h aP[4];
#pragma unroll
    for (int ks = 0; ks < 4; ++ks)
      aP[ks] = load_A(&Ps[wave][0][ks * 32], 136);
#pragma unroll
    for (int nd = 0; nd < 4; ++nd)
#pragma unroll
      for (int ks = 0; ks < 4; ++ks) {
        v16h bV = load_B(&Vts[buf][nd * 16][ks * 32], 136);
        oacc[nd] = wmma16(aP[ks], bV, oacc[nd]);
      }
  }

  // Normalize and store O as f16 [B,N,768].
#pragma unroll
  for (int nd = 0; nd < 4; ++nd)
#pragma unroll
    for (int r = 0; r < 8; ++r) {
      const int m = r + hf * 8;
      const int tok = qrow0 + m;
      const int d = nd * 16 + n16;
      const float val = oacc[nd][r] / row_l[r];
      obuf[((size_t)bb * SEQ + tok) * CDIM + hh * HDIM + d] = (f16)val;
    }
}

// ---------------------------------------------------------------------------
// Kernel 3: output projection. O(f16)[16384x768] x w_proj^T + bias -> f32 out.
// 256x128 block tile, 64x64 per wave.
// ---------------------------------------------------------------------------
__global__ __launch_bounds__(256) void proj_kernel(
    const f16* __restrict__ obuf, const float* __restrict__ wproj,
    const float* __restrict__ bias, float* __restrict__ out) {
  __shared__ f16 As[256][40];
  __shared__ f16 Bs[128][40];

  const int tid = threadIdx.x;
  const int rowBase = blockIdx.y * 256;
  const int colBase = blockIdx.x * 128;
  const int wave = tid >> 5;
  const int waveM = wave & 3, waveN = wave >> 2;

  v8f acc[4][4] = {};

  for (int kt = 0; kt < 24; ++kt) {
    const int k0 = kt * 32;
    __syncthreads();
#pragma unroll
    for (int t = 0; t < 4; ++t) {  // A: 256x32 f16 = 1024 x (8 f16)
      const int fl = tid + t * 256;
      const int row = fl >> 2, c8 = (fl & 3) << 3;
      *(uint4*)&As[row][c8] =
          *(const uint4*)(obuf + (size_t)(rowBase + row) * CDIM + k0 + c8);
    }
#pragma unroll
    for (int i = 0; i < 4; ++i) {  // B: convert f32 -> f16
      const int idx = tid + i * 256;
      const int row = idx >> 3, c4 = (idx & 7) << 2;
      float4 fb = *(const float4*)(wproj + (size_t)(colBase + row) * CDIM + k0 + c4);
      Bs[row][c4 + 0] = (f16)fb.x; Bs[row][c4 + 1] = (f16)fb.y;
      Bs[row][c4 + 2] = (f16)fb.z; Bs[row][c4 + 3] = (f16)fb.w;
    }
    __syncthreads();

    v16h a[4], b[4];
#pragma unroll
    for (int mt = 0; mt < 4; ++mt)
      a[mt] = load_A(&As[waveM * 64 + mt * 16][0], 40);
#pragma unroll
    for (int nt = 0; nt < 4; ++nt)
      b[nt] = load_B(&Bs[waveN * 64 + nt * 16][0], 40);
#pragma unroll
    for (int mt = 0; mt < 4; ++mt)
#pragma unroll
      for (int nt = 0; nt < 4; ++nt)
        acc[mt][nt] = wmma16(a[mt], b[nt], acc[mt][nt]);
  }

  const int lane = tid & 31, n16 = lane & 15, hf = lane >> 4;
#pragma unroll
  for (int mt = 0; mt < 4; ++mt)
#pragma unroll
    for (int nt = 0; nt < 4; ++nt)
#pragma unroll
      for (int r = 0; r < 8; ++r) {
        const int rowg = rowBase + waveM * 64 + mt * 16 + r + hf * 8;
        const int o = colBase + waveN * 64 + nt * 16 + n16;
        out[(size_t)rowg * CDIM + o] = acc[mt][nt][r] + bias[o];
      }
}

// ---------------------------------------------------------------------------
extern "C" void kernel_launch(void* const* d_in, const int* in_sizes, int n_in,
                              void* d_out, int out_size, void* d_ws,
                              size_t ws_size, hipStream_t stream) {
  const float* x      = (const float*)d_in[0];
  const float* w_qkv  = (const float*)d_in[1];
  const float* w_proj = (const float*)d_in[2];
  const float* b_proj = (const float*)d_in[3];
  float* out = (float*)d_out;

  const size_t HEAD_ELEMS = (size_t)BATCH * NHEAD * SEQ * HDIM;  // 12.58M f16
  f16* qws = (f16*)d_ws;
  f16* kws = qws + HEAD_ELEMS;
  f16* vws = kws + HEAD_ELEMS;
  f16* ows = vws + HEAD_ELEMS;  // [B, N, 768] f16

  qkv_kernel<<<dim3(3 * CDIM / 128, BATCH * SEQ / 256), 256, 0, stream>>>(
      x, w_qkv, qws, kws, vws);
  attn_kernel<<<dim3(SEQ / 128, BATCH * NHEAD), 256, 0, stream>>>(
      qws, kws, vws, ows);
  proj_kernel<<<dim3(CDIM / 128, BATCH * SEQ / 256), 256, 0, stream>>>(
      ows, w_proj, b_proj, out);
}
